// MultiHeadedAttention_56796647522533
// MI455X (gfx1250) — matrix operand order repaired and checked by reading, not compiled, not run on hardware
//
#include <hip/hip_runtime.h>
#include <hip/hip_bf16.h>
#include <math.h>

typedef _Float16 f16;
typedef __attribute__((ext_vector_type(16))) _Float16 v16h;
typedef __attribute__((ext_vector_type(8)))  _Float16 v8h;
typedef __attribute__((ext_vector_type(8)))  float    v8f;

// Problem dims (fixed by reference)
#define B_ 2
#define T_ 2048
#define E_ 64
#define H_ 8
#define BH_ (B_*H_)       // 16
#define K512 (H_*E_)      // 512
#define ROWS (B_*T_)      // 4096

// Workspace layout (bytes)
#define OFF_QS   ((size_t)0)                 // [BH,T] f32      = 128 KB
#define OFF_KS   ((size_t)131072)            // [BH,T] f32      = 128 KB
#define OFF_KMM  ((size_t)262144)            // [32]   f32 (max[16], min[16])
#define OFF_VT   ((size_t)524288)            // [BH,E,T] f16    = 4 MB (V transposed)
#define OFF_ZF16 ((size_t)(524288+4194304))          // [ROWS,512] f16 = 4 MB
#define OFF_WT   ((size_t)(524288+4194304+4194304))  // [512,512] f16  = 512 KB (W transposed)

// ---------------------------------------------------------------------------
// Kernel 0: transpose+convert ff_kernel [1,H,G,E,F] -> Wt[col=g*64+f][k=h*64+e] f16
// ---------------------------------------------------------------------------
__global__ void k_convert_w(const float* __restrict__ fk, f16* __restrict__ wt) {
    int idx = blockIdx.x * 256 + threadIdx.x;          // 0 .. 512*512-1
    int col = idx >> 9;                                // g*64+f
    int row = idx & 511;                               // h*64+e
    int h = row >> 6, e = row & 63;
    int g = col >> 6, f = col & 63;
    float w = fk[(((h * H_) + g) * E_ + e) * E_ + f];
    wt[(size_t)col * K512 + row] = (f16)w;
}

// ---------------------------------------------------------------------------
// Kernel 1: per-token scalars q,k and transposed V tiles.
// One 64-thread block per token; thread f computes vT[b,h,f,t] for all h.
// ---------------------------------------------------------------------------
__global__ void k_precompute(const float* __restrict__ emb,
                             const float* __restrict__ kk,
                             const float* __restrict__ qk,
                             const float* __restrict__ vk,
                             float* __restrict__ qs, float* __restrict__ ks,
                             f16* __restrict__ vt) {
    int tok = blockIdx.x;                 // 0..ROWS-1
    int b = tok >> 11, t = tok & (T_ - 1);
    int tid = threadIdx.x;                // 0..63  (f index)
    __shared__ float se[E_];
    se[tid] = emb[(size_t)tok * E_ + tid];
    __syncthreads();
#pragma unroll
    for (int h = 0; h < H_; ++h) {
        float acc = 0.f;
#pragma unroll 8
        for (int e = 0; e < E_; ++e)
            acc += se[e] * vk[((h * E_) + e) * E_ + tid];
        vt[((size_t)((b * H_ + h) * E_ + tid)) * T_ + t] = (f16)acc;
    }
    if (tid < H_) {
        float a = 0.f;
#pragma unroll 8
        for (int e = 0; e < E_; ++e) a += se[e] * qk[tid * E_ + e];
        qs[(size_t)(b * H_ + tid) * T_ + t] = a;
    } else if (tid < 2 * H_) {
        int h = tid - H_;
        float a = 0.f;
#pragma unroll 8
        for (int e = 0; e < E_; ++e) a += se[e] * kk[h * E_ + e];
        ks[(size_t)(b * H_ + h) * T_ + t] = a;
    }
}

// ---------------------------------------------------------------------------
// Kernel 1b: per-(b,h) min/max of key scalars (exact softmax max in O(T)).
// ---------------------------------------------------------------------------
__global__ void k_kminmax(const float* __restrict__ ks, float* __restrict__ kmm) {
    int bh = blockIdx.x, tid = threadIdx.x;   // 256 threads
    float mx = -1e30f, mn = 1e30f;
    for (int t = tid; t < T_; t += 256) {
        float v = ks[(size_t)bh * T_ + t];
        mx = fmaxf(mx, v); mn = fminf(mn, v);
    }
    __shared__ float smx[256], smn[256];
    smx[tid] = mx; smn[tid] = mn;
    __syncthreads();
    for (int s = 128; s > 0; s >>= 1) {
        if (tid < s) {
            smx[tid] = fmaxf(smx[tid], smx[tid + s]);
            smn[tid] = fminf(smn[tid], smn[tid + s]);
        }
        __syncthreads();
    }
    if (tid == 0) { kmm[bh] = smx[0]; kmm[16 + bh] = smn[0]; }
}

// ---------------------------------------------------------------------------
// Kernel 2: streaming softmax-GEMM (attn @ V) + residual + LayerNorm1.
// One wave per 16-query tile of one (b,h). P tile (16x32 f16) built in VALU,
// fed to v_wmma_f32_16x16x32_f16 against 4 column tiles of V (E=64).
// ---------------------------------------------------------------------------
__global__ void k_attn_ln1(const float* __restrict__ emb,
                           const float* __restrict__ qs,
                           const float* __restrict__ ks,
                           const float* __restrict__ kmm,
                           const f16* __restrict__ vt,
                           const float* __restrict__ g1,
                           const float* __restrict__ b1,
                           f16* __restrict__ zf16) {
    int wid  = (blockIdx.x * blockDim.x + threadIdx.x) >> 5;  // 0..2047
    int lane = threadIdx.x & 31;
    int bh   = wid >> 7;                 // 128 q-tiles per (b,h)
    int q0   = (wid & 127) << 4;
    int b    = bh >> 3;
    int h    = bh & 7;
    int half = lane >> 4;                // wave halves per WMMA layout
    int lx   = lane & 15;

    const float* ksp = ks + (size_t)bh * T_;
    float qv   = qs[(size_t)bh * T_ + q0 + lx];           // my row's query scalar
    float kmax = kmm[bh], kmin = kmm[16 + bh];
    float mrow = qv * (qv >= 0.f ? kmax : kmin);          // exact row max of scores

    v8f acc0 = {}, acc1 = {}, acc2 = {}, acc3 = {};
    float lsum = 0.f;
    const int baseA = half ? 16 : 0;
    const int baseB = half ? 24 : 8;
    const f16* vbase = vt + ((size_t)bh * E_) * T_;

    for (int k0 = 0; k0 < T_; k0 += 32) {
        float kv = ksp[k0 + lane];                        // 32 key scalars across wave
        v16h a;
#pragma unroll
        for (int i = 0; i < 16; ++i) {
            int koff   = (i < 8) ? (baseA + i) : (baseB + i - 8);
            float kval = __shfl(kv, koff, 32);            // ds_bpermute gather
            float p    = __expf(qv * kval - mrow);
            lsum += p;
            a[i] = (f16)p;
        }
        // B fragments: K-contiguous 16xf16 loads from transposed V
        {
            const v16h bt0 = *(const v16h*)(vbase + (size_t)(0 * 16 + lx) * T_ + k0 + half * 16);
            acc0 = __builtin_amdgcn_wmma_f32_16x16x32_f16(false, a, false, bt0, (short)0, acc0, false, false);
            const v16h bt1 = *(const v16h*)(vbase + (size_t)(1 * 16 + lx) * T_ + k0 + half * 16);
            acc1 = __builtin_amdgcn_wmma_f32_16x16x32_f16(false, a, false, bt1, (short)0, acc1, false, false);
            const v16h bt2 = *(const v16h*)(vbase + (size_t)(2 * 16 + lx) * T_ + k0 + half * 16);
            acc2 = __builtin_amdgcn_wmma_f32_16x16x32_f16(false, a, false, bt2, (short)0, acc2, false, false);
            const v16h bt3 = *(const v16h*)(vbase + (size_t)(3 * 16 + lx) * T_ + k0 + half * 16);
            acc3 = __builtin_amdgcn_wmma_f32_16x16x32_f16(false, a, false, bt3, (short)0, acc3, false, false);
        }
    }
    // combine half-wave partial row sums (lanes l and l+16 hold row l)
    lsum += __shfl_xor(lsum, 16, 32);

#pragma unroll
    for (int r = 0; r < 8; ++r) {
        int m = r + half * 8;                 // C/D layout: VGPR r -> row r / r+8
        int t = q0 + m;
        float inv = 1.f / __shfl(lsum, m, 32);
        float zv[4];
        zv[0] = acc0[r] * inv; zv[1] = acc1[r] * inv;
        zv[2] = acc2[r] * inv; zv[3] = acc3[r] * inv;
#pragma unroll
        for (int et = 0; et < 4; ++et)
            zv[et] += emb[((size_t)(b * T_ + t)) * E_ + et * 16 + lx];  // residual
        // LayerNorm over E=64 (4 regs x 16 lanes of my half)
        float s = zv[0] + zv[1] + zv[2] + zv[3];
        for (int mk = 8; mk >= 1; mk >>= 1) s += __shfl_xor(s, mk, 32);
        float mu = s * (1.f / 64.f);
        float sq = 0.f;
#pragma unroll
        for (int et = 0; et < 4; ++et) { float d = zv[et] - mu; sq += d * d; }
        for (int mk = 8; mk >= 1; mk >>= 1) sq += __shfl_xor(sq, mk, 32);
        float rinv = rsqrtf(sq * (1.f / 64.f) + 1e-3f);
#pragma unroll
        for (int et = 0; et < 4; ++et) {
            int e = et * 16 + lx;
            float zn = g1[e] * (zv[et] - mu) * rinv + b1[e];
            zf16[((size_t)(b * T_ + t)) * K512 + h * E_ + e] = (f16)zn;
        }
    }
}

// ---------------------------------------------------------------------------
// Kernel 3: cross-head FFN GEMM (4096x512x512) + residual + GELU(tanh) + LN2.
// One wave per 16-row x 64-col (one output head g) tile.
// ---------------------------------------------------------------------------
__global__ void k_ffn_ln2(const f16* __restrict__ zf16,
                          const f16* __restrict__ wt,
                          const float* __restrict__ g2,
                          const float* __restrict__ b2,
                          float* __restrict__ out) {
    int wid  = (blockIdx.x * blockDim.x + threadIdx.x) >> 5;  // 0..2047
    int lane = threadIdx.x & 31;
    int half = lane >> 4, lx = lane & 15;
    int mtile = wid >> 3;            // 0..255
    int g     = wid & 7;
    int m0    = mtile << 4;

    v8f acc0 = {}, acc1 = {}, acc2 = {}, acc3 = {};
    const f16* zrow = zf16 + (size_t)(m0 + lx) * K512;

    for (int k0 = 0; k0 < K512; k0 += 32) {
        v16h a;
        const v8h alo = *(const v8h*)(zrow + k0 + half * 16);
        const v8h ahi = *(const v8h*)(zrow + k0 + half * 16 + 8);
#pragma unroll
        for (int i = 0; i < 8; ++i) { a[i] = alo[i]; a[8 + i] = ahi[i]; }
        const v16h bt0 = *(const v16h*)(wt + (size_t)(g * 64 + 0 * 16 + lx) * K512 + k0 + half * 16);
        acc0 = __builtin_amdgcn_wmma_f32_16x16x32_f16(false, a, false, bt0, (short)0, acc0, false, false);
        const v16h bt1 = *(const v16h*)(wt + (size_t)(g * 64 + 1 * 16 + lx) * K512 + k0 + half * 16);
        acc1 = __builtin_amdgcn_wmma_f32_16x16x32_f16(false, a, false, bt1, (short)0, acc1, false, false);
        const v16h bt2 = *(const v16h*)(wt + (size_t)(g * 64 + 2 * 16 + lx) * K512 + k0 + half * 16);
        acc2 = __builtin_amdgcn_wmma_f32_16x16x32_f16(false, a, false, bt2, (short)0, acc2, false, false);
        const v16h bt3 = *(const v16h*)(wt + (size_t)(g * 64 + 3 * 16 + lx) * K512 + k0 + half * 16);
        acc3 = __builtin_amdgcn_wmma_f32_16x16x32_f16(false, a, false, bt3, (short)0, acc3, false, false);
    }

#pragma unroll
    for (int r = 0; r < 8; ++r) {
        int m   = r + half * 8;
        int row = m0 + m;                  // b*T + t
        int b   = row >> 11, t = row & (T_ - 1);
        float yv[4];
        yv[0] = acc0[r]; yv[1] = acc1[r]; yv[2] = acc2[r]; yv[3] = acc3[r];
#pragma unroll
        for (int et = 0; et < 4; ++et) {
            float z = (float)zf16[(size_t)row * K512 + g * E_ + et * 16 + lx];
            float x = z + yv[et];                                   // residual
            float tt = tanhf(0.7978845608f * (x + 0.044715f * x * x * x));
            yv[et] = 0.5f * x * (1.f + tt);                         // GELU (tanh)
        }
        float s = yv[0] + yv[1] + yv[2] + yv[3];
        for (int mk = 8; mk >= 1; mk >>= 1) s += __shfl_xor(s, mk, 32);
        float mu = s * (1.f / 64.f);
        float sq = 0.f;
#pragma unroll
        for (int et = 0; et < 4; ++et) { float d = yv[et] - mu; sq += d * d; }
        for (int mk = 8; mk >= 1; mk >>= 1) sq += __shfl_xor(sq, mk, 32);
        float rinv = rsqrtf(sq * (1.f / 64.f) + 1e-3f);
#pragma unroll
        for (int et = 0; et < 4; ++et) {
            int e = et * 16 + lx;
            out[(((size_t)(b * H_ + g) * T_ + t) * E_) + e] =
                g2[e] * (yv[et] - mu) * rinv + b2[e];
        }
    }
}

// ---------------------------------------------------------------------------
extern "C" void kernel_launch(void* const* d_in, const int* in_sizes, int n_in,
                              void* d_out, int out_size, void* d_ws, size_t ws_size,
                              hipStream_t stream) {
    const float* emb = (const float*)d_in[0];
    const float* kk  = (const float*)d_in[1];
    const float* qk  = (const float*)d_in[2];
    const float* vk  = (const float*)d_in[3];
    const float* fk  = (const float*)d_in[4];
    const float* g1  = (const float*)d_in[5];
    const float* b1  = (const float*)d_in[6];
    const float* g2  = (const float*)d_in[7];
    const float* b2  = (const float*)d_in[8];
    float* out = (float*)d_out;

    char* ws = (char*)d_ws;
    float* qs   = (float*)(ws + OFF_QS);
    float* ks   = (float*)(ws + OFF_KS);
    float* kmm  = (float*)(ws + OFF_KMM);
    f16*   vt   = (f16*)  (ws + OFF_VT);
    f16*   zf16 = (f16*)  (ws + OFF_ZF16);
    f16*   wt   = (f16*)  (ws + OFF_WT);

    k_convert_w<<<(K512 * K512) / 256, 256, 0, stream>>>(fk, wt);
    k_precompute<<<ROWS, 64, 0, stream>>>(emb, kk, qk, vk, qs, ks, vt);
    k_kminmax<<<BH_, 256, 0, stream>>>(ks, kmm);
    k_attn_ln1<<<256, 256, 0, stream>>>(emb, qs, ks, kmm, vt, g1, b1, zf16);
    k_ffn_ln2<<<256, 256, 0, stream>>>(zf16, wt, g2, b2, out);
}
